// Transformer_54073638256771
// MI455X (gfx1250) — matrix operand order repaired and checked by reading, not compiled, not run on hardware
//
#include <hip/hip_runtime.h>

typedef __bf16 bf16;
typedef __attribute__((ext_vector_type(16))) __bf16 v16bf;
typedef __attribute__((ext_vector_type(8)))  __bf16 v8bf;
typedef __attribute__((ext_vector_type(8)))  float  v8f;

constexpr int B_ = 4, S_ = 512, D_ = 512, H_ = 8, L_ = 6;
constexpr int DFF_ = 2048, V_ = 32000, DK_ = 64;
constexpr int MR_ = B_ * S_;  // 2048 rows of activations

// ---------------------------------------------------------------------------
// Fragment loaders (contiguous k order, not the ISA table order; 16-bit A 16x32 / B 32x16)
// A (16x32): lanes 0-15 hold K=[0..15]; lanes 16-31 hold K=[16..31] (contiguous)
// B (32x16): lanes 0-15 hold K=[0..15]; lanes 16-31 hold K=[16..31] (contiguous)
// ---------------------------------------------------------------------------
__device__ __forceinline__ v16bf ldA(const bf16* rowp, int k, int hi) {
  return *(const v16bf*)(rowp + k + hi * 16);
}
__device__ __forceinline__ v16bf ldB(const bf16* rowp, int k, int hi) {
  return *(const v16bf*)(rowp + k + hi * 16);
}

// ---------------------------------------------------------------------------
// Batched bf16 GEMM: C[MxN] = A[MxK] * Bt[NxK]^T, alpha/bias/relu epilogue.
// Block = 8 waves arranged WR(M) x WC(N); each wave owns a (WMT*16)x(WNT*16)
// tile = WMT x WNT WMMA tiles. Fragments are reused WNT/WMT times each, so
// arithmetic intensity of the inner loop scales with the tile area.
// Batch index z decomposed as (zb, zh) = (z / Hb, z % Hb) with 2-level strides.
// ---------------------------------------------------------------------------
template <int WMT, int WNT, int WR, int WC>
__global__ __launch_bounds__(256)
void gemm_bf16_kernel(const bf16* __restrict__ A, int lda, long long sAb, long long sAh,
                      const bf16* __restrict__ Bt, int ldb, long long sBb, long long sBh,
                      float* __restrict__ C, bf16* __restrict__ Cbf,
                      int ldc, long long sCb, long long sCh,
                      const float* __restrict__ bias,
                      int M, int N, int K, int Hb, float alpha, int relu) {
  int z = blockIdx.z;
  int zb = z / Hb, zh = z - zb * Hb;
  A  += (long long)zb * sAb + (long long)zh * sAh;
  Bt += (long long)zb * sBb + (long long)zh * sBh;
  long long co = (long long)zb * sCb + (long long)zh * sCh;

  int wave = threadIdx.x >> 5;
  int lane = threadIdx.x & 31;
  int wm = wave % WR, wn = wave / WR;
  int row0 = blockIdx.y * (WR * WMT * 16) + wm * (WMT * 16);
  int col0 = blockIdx.x * (WC * WNT * 16) + wn * (WNT * 16);
  if (row0 >= M || col0 >= N) return;

  int mi = lane & 15;
  int hi = lane >> 4;

  const bf16* ap[WMT];
  const bf16* bp[WNT];
#pragma unroll
  for (int i = 0; i < WMT; ++i) ap[i] = A + (long long)(row0 + i * 16 + mi) * lda;
#pragma unroll
  for (int j = 0; j < WNT; ++j) bp[j] = Bt + (long long)(col0 + j * 16 + mi) * ldb;

  v8f acc[WMT][WNT] = {};
  for (int k = 0; k < K; k += 32) {
    v16bf af[WMT], bfr[WNT];
#pragma unroll
    for (int i = 0; i < WMT; ++i) af[i] = ldA(ap[i], k, hi);
#pragma unroll
    for (int j = 0; j < WNT; ++j) bfr[j] = ldB(bp[j], k, hi);
#pragma unroll
    for (int i = 0; i < WMT; ++i)
#pragma unroll
      for (int j = 0; j < WNT; ++j)
        acc[i][j] = __builtin_amdgcn_wmma_f32_16x16x32_bf16(
            false, af[i], false, bfr[j], (short)0, acc[i][j], false, false);
  }

  int cn = lane & 15;
  int rb = hi * 8;  // C/D: VGPR r -> M = r (lanes 0-15), M = r+8 (lanes 16-31)
#pragma unroll
  for (int i = 0; i < WMT; ++i) {
#pragma unroll
    for (int j = 0; j < WNT; ++j) {
#pragma unroll
      for (int r = 0; r < 8; ++r) {
        int row = row0 + i * 16 + rb + r;
        int col = col0 + j * 16 + cn;
        float v = acc[i][j][r] * alpha;
        if (bias) v += bias[col];
        if (relu) v = fmaxf(v, 0.f);
        long long idx = co + (long long)row * ldc + col;
        if (C)   C[idx]   = v;
        if (Cbf) Cbf[idx] = (bf16)v;
      }
    }
  }
}

// fp32 [K,N] row-major -> bf16 [N,K] row-major (weight transpose+convert)
__global__ __launch_bounds__(256)
void convert_wt_kernel(const float* __restrict__ W, bf16* __restrict__ Wt, int K, int N) {
  __shared__ float tile[32][33];
  int nb = blockIdx.x * 32, kb = blockIdx.y * 32;
  int tx = threadIdx.x, ty = threadIdx.y;
  for (int i = ty; i < 32; i += 8) tile[i][tx] = W[(long long)(kb + i) * N + nb + tx];
  __syncthreads();
  for (int i = ty; i < 32; i += 8) Wt[(long long)(nb + i) * K + kb + tx] = (bf16)tile[tx][i];
}

// V [MR,D] bf16 (heads packed in cols) -> Vt [B*H][DK][S] bf16
__global__ __launch_bounds__(256)
void transpose_v_kernel(const bf16* __restrict__ Vb, bf16* __restrict__ Vt) {
  __shared__ bf16 tile[32][33];
  int bh = blockIdx.z, b = bh / H_, h = bh - b * H_;
  int sb = blockIdx.x * 32, db = blockIdx.y * 32;
  int tx = threadIdx.x, ty = threadIdx.y;
  for (int i = ty; i < 32; i += 8)
    tile[i][tx] = Vb[(long long)(b * S_ + sb + i) * D_ + h * DK_ + db + tx];
  __syncthreads();
  for (int i = ty; i < 32; i += 8)
    Vt[((long long)bh * DK_ + db + i) * S_ + sb + tx] = tile[tx][i];
}

// row softmax over keys, optional causal mask; writes bf16 probabilities
__global__ __launch_bounds__(256)
void softmax_attn_kernel(const float* __restrict__ scores, bf16* __restrict__ P, int causal) {
  int bh = blockIdx.y, i = blockIdx.x, t = threadIdx.x;
  const float* row = scores + ((long long)bh * S_ + i) * S_;
  bf16* prow = P + ((long long)bh * S_ + i) * S_;
  int jmax = causal ? i : (S_ - 1);
  __shared__ float red[256];
  float lmax = -INFINITY;
  for (int j = t; j < S_; j += 256) if (j <= jmax) lmax = fmaxf(lmax, row[j]);
  red[t] = lmax; __syncthreads();
  for (int s = 128; s > 0; s >>= 1) { if (t < s) red[t] = fmaxf(red[t], red[t + s]); __syncthreads(); }
  float m = red[0]; __syncthreads();
  float lsum = 0.f;
  for (int j = t; j < S_; j += 256) if (j <= jmax) lsum += __expf(row[j] - m);
  red[t] = lsum; __syncthreads();
  for (int s = 128; s > 0; s >>= 1) { if (t < s) red[t] += red[t + s]; __syncthreads(); }
  float inv = 1.f / red[0];
  for (int j = t; j < S_; j += 256)
    prow[j] = (bf16)((j <= jmax) ? __expf(row[j] - m) * inv : 0.f);
}

// out = LayerNorm(x + d) * g + b; writes fp32 + bf16 mirror (D = 512, in place OK)
__global__ __launch_bounds__(256)
void add_ln_kernel(const float* __restrict__ x, const float* __restrict__ d,
                   const float* __restrict__ g, const float* __restrict__ b,
                   float* __restrict__ outf, bf16* __restrict__ outb) {
  int row = blockIdx.x, t = threadIdx.x;
  long long base = (long long)row * D_;
  float v0 = x[base + t] + d[base + t];
  float v1 = x[base + t + 256] + d[base + t + 256];
  __shared__ float red[256];
  red[t] = v0 + v1; __syncthreads();
  for (int s = 128; s > 0; s >>= 1) { if (t < s) red[t] += red[t + s]; __syncthreads(); }
  float mean = red[0] * (1.f / D_); __syncthreads();
  float d0 = v0 - mean, d1 = v1 - mean;
  red[t] = d0 * d0 + d1 * d1; __syncthreads();
  for (int s = 128; s > 0; s >>= 1) { if (t < s) red[t] += red[t + s]; __syncthreads(); }
  float rstd = rsqrtf(red[0] * (1.f / D_) + 1e-5f);
  float o0 = d0 * rstd * g[t] + b[t];
  float o1 = d1 * rstd * g[t + 256] + b[t + 256];
  outf[base + t] = o0;        outf[base + t + 256] = o1;
  outb[base + t] = (bf16)o0;  outb[base + t + 256] = (bf16)o1;
}

// h = emb[token]*sqrt(D) + PE; writes fp32 + bf16 mirror
__global__ __launch_bounds__(256)
void embed_kernel(const int* __restrict__ ids, const float* __restrict__ emb,
                  float* __restrict__ outf, bf16* __restrict__ outb) {
  int row = blockIdx.x;
  int s = row % S_;
  int tok = ids[row];
  const float* e = emb + (long long)tok * D_;
  const float scale = 22.62741699796952f;          // sqrt(512)
  const float neglogn = -9.210340371976184f;       // -ln(10000)
  for (int dd = threadIdx.x; dd < D_; dd += 256) {
    float freq = __expf((float)(dd & ~1) * (neglogn / (float)D_));
    float ang = (float)s * freq;
    float pe = (dd & 1) ? __cosf(ang) : __sinf(ang);
    float v = e[dd] * scale + pe;
    long long idx = (long long)row * D_ + dd;
    outf[idx] = v;
    outb[idx] = (bf16)v;
  }
}

// softmax over the SEQUENCE axis (dim=1 of [B,S,V]), in place
__global__ __launch_bounds__(256)
void softmax_seq_kernel(float* __restrict__ logits) {
  int b = blockIdx.y;
  int vo = threadIdx.x & 63;
  int sg = threadIdx.x >> 6;  // 0..3
  int v = blockIdx.x * 64 + vo;
  float* base = logits + (long long)b * S_ * V_ + v;
  __shared__ float red[4][64];
  float lmax = -INFINITY;
  for (int s = sg; s < S_; s += 4) lmax = fmaxf(lmax, base[(long long)s * V_]);
  red[sg][vo] = lmax; __syncthreads();
  float m = fmaxf(fmaxf(red[0][vo], red[1][vo]), fmaxf(red[2][vo], red[3][vo]));
  __syncthreads();
  float lsum = 0.f;
  for (int s = sg; s < S_; s += 4) lsum += __expf(base[(long long)s * V_] - m);
  red[sg][vo] = lsum; __syncthreads();
  float inv = 1.f / (red[0][vo] + red[1][vo] + red[2][vo] + red[3][vo]);
  for (int s = sg; s < S_; s += 4) {
    long long idx = (long long)s * V_;
    base[idx] = __expf(base[idx] - m) * inv;
  }
}

// ---------------------------------------------------------------------------
extern "C" void kernel_launch(void* const* d_in, const int* in_sizes, int n_in,
                              void* d_out, int out_size, void* d_ws, size_t ws_size,
                              hipStream_t stream) {
  (void)in_sizes; (void)n_in; (void)out_size; (void)ws_size;
  const int*   x          = (const int*)d_in[0];
  const int*   y          = (const int*)d_in[1];
  const float* emb_in     = (const float*)d_in[2];
  const float* emb_out    = (const float*)d_in[3];
  const float* enc_attn_w = (const float*)d_in[4];
  const float* enc_attn_b = (const float*)d_in[5];
  const float* enc_ln_g   = (const float*)d_in[6];
  const float* enc_ln_b   = (const float*)d_in[7];
  const float* enc_ffn_w1 = (const float*)d_in[8];
  const float* enc_ffn_b1 = (const float*)d_in[9];
  const float* enc_ffn_w2 = (const float*)d_in[10];
  const float* enc_ffn_b2 = (const float*)d_in[11];
  const float* dec_sa_w   = (const float*)d_in[12];
  const float* dec_sa_b   = (const float*)d_in[13];
  const float* dec_ca_w   = (const float*)d_in[14];
  const float* dec_ca_b   = (const float*)d_in[15];
  const float* dec_ln_g   = (const float*)d_in[16];
  const float* dec_ln_b   = (const float*)d_in[17];
  const float* dec_ffn_w1 = (const float*)d_in[18];
  const float* dec_ffn_b1 = (const float*)d_in[19];
  const float* dec_ffn_w2 = (const float*)d_in[20];
  const float* dec_ffn_b2 = (const float*)d_in[21];
  const float* out_w      = (const float*)d_in[22];
  const float* out_b      = (const float*)d_in[23];

  // ---- workspace carving -------------------------------------------------
  size_t off = 0;
  auto carve = [&](size_t bytes) -> void* {
    void* p = (char*)d_ws + off;
    off += (bytes + 255) & ~(size_t)255;
    return p;
  };
  bf16* encAttnWt = (bf16*)carve((size_t)L_ * 4 * D_ * D_ * 2);
  bf16* encW1t    = (bf16*)carve((size_t)L_ * DFF_ * D_ * 2);
  bf16* encW2t    = (bf16*)carve((size_t)L_ * D_ * DFF_ * 2);
  bf16* decSaWt   = (bf16*)carve((size_t)L_ * 4 * D_ * D_ * 2);
  bf16* decCaWt   = (bf16*)carve((size_t)L_ * 4 * D_ * D_ * 2);
  bf16* decW1t    = (bf16*)carve((size_t)L_ * DFF_ * D_ * 2);
  bf16* decW2t    = (bf16*)carve((size_t)L_ * D_ * DFF_ * 2);
  bf16* outWt     = (bf16*)carve((size_t)V_ * D_ * 2);
  float* hF       = (float*)carve((size_t)MR_ * D_ * 4);
  float* delta    = (float*)carve((size_t)MR_ * D_ * 4);
  bf16* hB        = (bf16*)carve((size_t)MR_ * D_ * 2);
  bf16* encB      = (bf16*)carve((size_t)MR_ * D_ * 2);
  bf16* qB        = (bf16*)carve((size_t)MR_ * D_ * 2);
  bf16* kB        = (bf16*)carve((size_t)MR_ * D_ * 2);
  bf16* vB        = (bf16*)carve((size_t)MR_ * D_ * 2);
  bf16* vtB       = (bf16*)carve((size_t)MR_ * D_ * 2);
  bf16* oB        = (bf16*)carve((size_t)MR_ * D_ * 2);
  float* scoresF  = (float*)carve((size_t)B_ * H_ * S_ * S_ * 4);
  bf16* pB        = (bf16*)carve((size_t)B_ * H_ * S_ * S_ * 2);
  bf16* fB        = (bf16*)carve((size_t)MR_ * DFF_ * 2);

  // ---- one-time (per call) weight convert+transpose to bf16 --------------
  dim3 cb(32, 8);
  auto conv = [&](const float* W, bf16* Wt, int K, int N) {
    convert_wt_kernel<<<dim3(N / 32, K / 32), cb, 0, stream>>>(W, Wt, K, N);
  };
  for (int m = 0; m < L_ * 4; ++m) {
    conv(enc_attn_w + (size_t)m * D_ * D_, encAttnWt + (size_t)m * D_ * D_, D_, D_);
    conv(dec_sa_w   + (size_t)m * D_ * D_, decSaWt   + (size_t)m * D_ * D_, D_, D_);
    conv(dec_ca_w   + (size_t)m * D_ * D_, decCaWt   + (size_t)m * D_ * D_, D_, D_);
  }
  for (int i = 0; i < L_; ++i) {
    conv(enc_ffn_w1 + (size_t)i * D_ * DFF_, encW1t + (size_t)i * DFF_ * D_, D_, DFF_);
    conv(enc_ffn_w2 + (size_t)i * DFF_ * D_, encW2t + (size_t)i * D_ * DFF_, DFF_, D_);
    conv(dec_ffn_w1 + (size_t)i * D_ * DFF_, decW1t + (size_t)i * DFF_ * D_, D_, DFF_);
    conv(dec_ffn_w2 + (size_t)i * DFF_ * D_, decW2t + (size_t)i * D_ * DFF_, DFF_, D_);
  }
  conv(out_w, outWt, D_, V_);

  // ---- GEMM launchers ----------------------------------------------------
  // BIG: 8 waves as 2(M)x4(N), 64x64 per wave -> 128x256 block, 16 WMMA / k-step
  auto gemmBig = [&](const bf16* A, int lda, long long sAb, long long sAh,
                     const bf16* Bt, int ldb, long long sBb, long long sBh,
                     float* C, bf16* Cbf, int ldc, long long sCb, long long sCh,
                     const float* bias, int M, int N, int K, int batch, int Hb,
                     float alpha, int relu) {
    dim3 g((N + 255) / 256, (M + 127) / 128, batch);
    gemm_bf16_kernel<4, 4, 2, 4><<<g, 256, 0, stream>>>(
        A, lda, sAb, sAh, Bt, ldb, sBb, sBh, C, Cbf, ldc, sCb, sCh, bias,
        M, N, K, Hb, alpha, relu);
  };
  // SMALL: 8 waves as 4(M)x2(N), 32x32 per wave -> 128x64 block (for N=64)
  auto gemmSmall = [&](const bf16* A, int lda, long long sAb, long long sAh,
                       const bf16* Bt, int ldb, long long sBb, long long sBh,
                       float* C, bf16* Cbf, int ldc, long long sCb, long long sCh,
                       const float* bias, int M, int N, int K, int batch, int Hb,
                       float alpha, int relu) {
    dim3 g((N + 63) / 64, (M + 127) / 128, batch);
    gemm_bf16_kernel<2, 2, 4, 2><<<g, 256, 0, stream>>>(
        A, lda, sAb, sAh, Bt, ldb, sBb, sBh, C, Cbf, ldc, sCb, sCh, bias,
        M, N, K, Hb, alpha, relu);
  };

  const float inv_sqrt_d = 0.044194173824159216f;  // 1/sqrt(512) (NB: d_model)

  auto mha = [&](const bf16* xq, const bf16* xkv, const bf16* wt4, const float* b4,
                 int causal) {
    // Q, K, V projections (bf16 outputs only)
    gemmBig(xq,  D_, 0, 0, wt4 + 0 * (size_t)D_ * D_, D_, 0, 0, nullptr, qB, D_, 0, 0,
            b4 + 0 * D_, MR_, D_, D_, 1, 1, 1.f, 0);
    gemmBig(xkv, D_, 0, 0, wt4 + 1 * (size_t)D_ * D_, D_, 0, 0, nullptr, kB, D_, 0, 0,
            b4 + 1 * D_, MR_, D_, D_, 1, 1, 1.f, 0);
    gemmBig(xkv, D_, 0, 0, wt4 + 2 * (size_t)D_ * D_, D_, 0, 0, nullptr, vB, D_, 0, 0,
            b4 + 2 * D_, MR_, D_, D_, 1, 1, 1.f, 0);
    transpose_v_kernel<<<dim3(S_ / 32, DK_ / 32, B_ * H_), dim3(32, 8), 0, stream>>>(vB, vtB);
    // scores = Q Kt / sqrt(D): per-(b,h) strided views, K used directly as [N,K] operand
    gemmBig(qB, D_, (long long)S_ * D_, DK_,
            kB, D_, (long long)S_ * D_, DK_,
            scoresF, nullptr, S_, (long long)H_ * S_ * S_, (long long)S_ * S_,
            nullptr, S_, S_, DK_, B_ * H_, H_, inv_sqrt_d, 0);
    softmax_attn_kernel<<<dim3(S_, B_ * H_), 256, 0, stream>>>(scoresF, pB, causal);
    // O = P V: Vt is the [N=DK, K=S] operand; output written into merged-head layout
    gemmSmall(pB, S_, (long long)H_ * S_ * S_, (long long)S_ * S_,
              vtB, S_, (long long)H_ * DK_ * S_, (long long)DK_ * S_,
              nullptr, oB, D_, (long long)S_ * D_, DK_,
              nullptr, S_, DK_, S_, B_ * H_, H_, 1.f, 0);
    // output projection -> fp32 delta (for residual+LN)
    gemmBig(oB, D_, 0, 0, wt4 + 3 * (size_t)D_ * D_, D_, 0, 0, delta, nullptr, D_, 0, 0,
            b4 + 3 * D_, MR_, D_, D_, 1, 1, 1.f, 0);
  };

  auto ffn = [&](const bf16* w1t, const float* b1, const bf16* w2t, const float* b2) {
    gemmBig(hB, D_, 0, 0, w1t, D_, 0, 0, nullptr, fB, DFF_, 0, 0, b1,
            MR_, DFF_, D_, 1, 1, 1.f, 1);  // ReLU fused
    gemmBig(fB, DFF_, 0, 0, w2t, DFF_, 0, 0, delta, nullptr, D_, 0, 0, b2,
            MR_, D_, DFF_, 1, 1, 1.f, 0);
  };

  auto addln = [&](const float* g, const float* b) {
    add_ln_kernel<<<MR_, 256, 0, stream>>>(hF, delta, g, b, hF, hB);
  };

  // ---- encoder -----------------------------------------------------------
  embed_kernel<<<MR_, 256, 0, stream>>>(x, emb_in, hF, hB);
  for (int i = 0; i < L_; ++i) {
    mha(hB, hB, encAttnWt + (size_t)i * 4 * D_ * D_, enc_attn_b + (size_t)i * 4 * D_, 0);
    addln(enc_ln_g + (size_t)(i * 2 + 0) * D_, enc_ln_b + (size_t)(i * 2 + 0) * D_);
    ffn(encW1t + (size_t)i * DFF_ * D_, enc_ffn_b1 + (size_t)i * DFF_,
        encW2t + (size_t)i * D_ * DFF_, enc_ffn_b2 + (size_t)i * D_);
    addln(enc_ln_g + (size_t)(i * 2 + 1) * D_, enc_ln_b + (size_t)(i * 2 + 1) * D_);
  }
  hipMemcpyAsync(encB, hB, (size_t)MR_ * D_ * sizeof(bf16), hipMemcpyDeviceToDevice, stream);

  // ---- decoder -----------------------------------------------------------
  embed_kernel<<<MR_, 256, 0, stream>>>(y, emb_out, hF, hB);
  for (int i = 0; i < L_; ++i) {
    mha(hB, hB, decSaWt + (size_t)i * 4 * D_ * D_, dec_sa_b + (size_t)i * 4 * D_, 1);
    addln(dec_ln_g + (size_t)(i * 3 + 0) * D_, dec_ln_b + (size_t)(i * 3 + 0) * D_);
    mha(hB, encB, decCaWt + (size_t)i * 4 * D_ * D_, dec_ca_b + (size_t)i * 4 * D_, 0);
    addln(dec_ln_g + (size_t)(i * 3 + 1) * D_, dec_ln_b + (size_t)(i * 3 + 1) * D_);
    ffn(decW1t + (size_t)i * DFF_ * D_, dec_ffn_b1 + (size_t)i * DFF_,
        decW2t + (size_t)i * D_ * DFF_, dec_ffn_b2 + (size_t)i * D_);
    addln(dec_ln_g + (size_t)(i * 3 + 2) * D_, dec_ln_b + (size_t)(i * 3 + 2) * D_);
  }

  // ---- logits into d_out, then softmax over sequence axis (dim=1) --------
  gemmBig(hB, D_, 0, 0, outWt, D_, 0, 0, (float*)d_out, nullptr, V_, 0, 0, out_b,
          MR_, V_, D_, 1, 1, 1.f, 0);
  softmax_seq_kernel<<<dim3(V_ / 64, B_), 256, 0, stream>>>((float*)d_out);
}
